// QNetwork_42296837931298
// MI455X (gfx1250) — matrix operand order repaired and checked by reading, not compiled, not run on hardware
//
#include <hip/hip_runtime.h>
#include <hip/hip_bf16.h>

typedef __attribute__((ext_vector_type(16))) _Float16 v16h;
typedef __attribute__((ext_vector_type(8)))  float    v8f;
typedef int vi4 __attribute__((vector_size(16)));

#define Bn   512
#define Fch  4
#define Op   10
#define Hh   128
#define Ww   128
#define BASEc 64
#define OUTc  5

#if defined(__gfx1250__) && __has_builtin(__builtin_amdgcn_global_load_async_to_lds_b128)
#define HAVE_ASYNC_LDS 1
#else
#define HAVE_ASYNC_LDS 0
#endif

// ---------------------------------------------------------------------------
// 1) Rasterize: one block per (b,f) image, built in LDS, ordered markers by t0
// ---------------------------------------------------------------------------
__global__ __launch_bounds__(256) void raster_kernel(const int* __restrict__ x,
                                                     _Float16* __restrict__ t) {
    __shared__ __align__(16) _Float16 sh[Hh * Ww];
    const int bf = blockIdx.x;                 // 0 .. B*F-1
    const int tid = threadIdx.x;
    const int* xb = x + (size_t)bf * Op * 2;

    const int d0 = xb[(Op - 1) * 2 + 0];
    const int d1 = xb[(Op - 1) * 2 + 1];
    const int dX = BASEc + d0;
    const int dY = BASEc - d1;
    const int loX = (dX >= BASEc) ? BASEc : dX + 1;
    const int hiX = (dX >= BASEc) ? dX - 1 : BASEc;
    const int loY = (dY >= BASEc) ? BASEc : dY + 1;
    const int hiY = (dY >= BASEc) ? dY - 1 : BASEc;

    for (int i = tid; i < Hh * Ww; i += 256) {
        const int y  = i >> 7;
        const int xp = i & 127;
        float v = 0.f;
        const bool px = (xp >= loX) && (xp <= hiX) && (y == BASEc);
        const bool py = (y >= loY) && (y <= hiY) && (xp == dX);
        if (px || py) v = -1.f;
        if (y >= BASEc - 1 && y <= BASEc + 1 && xp >= BASEc - 1 && xp <= BASEc + 1) v = 0.5f;
        if (y == BASEc && xp == BASEc) v = 1.f;
        sh[i] = (_Float16)v;
    }
    __syncthreads();

    if (tid == 0) {
        // sequential marker pass (later points overwrite earlier ones)
        for (int k = 0; k < Op - 1; ++k) {
            int a  = ((xb[k * 2 + 0] + BASEc) % (Ww - 1) + (Ww - 1)) % (Ww - 1);
            int bb = ((BASEc - xb[k * 2 + 1]) % (Hh - 1) + (Hh - 1)) % (Hh - 1);
            for (int dn = -1; dn <= 1; ++dn)
                for (int dm = -1; dm <= 1; ++dm) {
                    int yy = bb + dn, xx = a + dm;
                    if (yy >= 0 && yy < Hh && xx >= 0 && xx < Ww)
                        sh[yy * Ww + xx] = (_Float16)0.5f;
                }
            sh[bb * Ww + a] = (_Float16)1.0f;
        }
    }
    __syncthreads();

    // vectorized writeout: 16384 halves = 2048 uint4
    uint4* dst = (uint4*)(t + (size_t)bf * Hh * Ww);
    const uint4* src = (const uint4*)sh;
    for (int i = tid; i < (Hh * Ww) / 8; i += 256) dst[i] = src[i];
}

// ---------------------------------------------------------------------------
// 2) Pack weights [Cout,K] f32 -> per-lane WMMA A layout, zero-padded k-steps
//    pw[((mt*Ksteps+s)*32 + lane)*16 + e], lane: m=lane%16, h=lane/16
//    e<8 : k = s*32 + 8h + e ;  e>=8 : k = s*32 + 16 + 8h + (e-8)
// ---------------------------------------------------------------------------
__global__ void pack_weights(const float* __restrict__ w, _Float16* __restrict__ pw,
                             int Cout, int K, int Ksteps, int mTiles) {
    int idx = blockIdx.x * blockDim.x + threadIdx.x;
    int total = mTiles * Ksteps * 512;
    if (idx >= total) return;
    int e    = idx & 15;
    int lane = (idx >> 4) & 31;
    int s    = (idx >> 9) % Ksteps;
    int mt   = (idx >> 9) / Ksteps;
    int m  = lane & 15;
    int hh = lane >> 4;
    int kin = (e < 8) ? (8 * hh + e) : (16 + 8 * hh + (e - 8));
    int k   = s * 32 + kin;
    int co  = mt * 16 + m;
    float v = (k < K && co < Cout) ? w[(size_t)co * K + k] : 0.f;
    pw[idx] = (_Float16)v;
}

// ---------------------------------------------------------------------------
// 3) LDS-staged implicit-GEMM conv via v_wmma_f32_16x16x32_f16
//    One workgroup (8 waves) per batch image. The whole Cin*Hin*Win f16 image
//    is staged into LDS once (async global->LDS: ASYNCcnt + s_wait_asynccnt),
//    so each input element is read from HBM exactly once — the bandwidth-
//    roofline minimum — and all im2col gathers become ds_load.
//    Each wave owns 16-position n-tiles (strided by 8) and ALL MT m-tiles.
// ---------------------------------------------------------------------------
template <int Cin, int Hin, int Win, int Cout, int Hout, int Wout,
          int KH, int KW, int STR, int PAD, int KSTEPS, int MT>
__global__ __launch_bounds__(256) void conv_wmma(
    const _Float16* __restrict__ in, const _Float16* __restrict__ pw,
    const float* __restrict__ bias, _Float16* __restrict__ out)
{
    extern __shared__ _Float16 sh[];          // Cin*Hin*Win halves
    constexpr int IMG   = Cin * Hin * Win;    // halves (divisible by 8)
    constexpr int P     = Hout * Wout;        // positions per image
    constexpr int TILES = (P + 15) / 16;
    constexpr int K     = Cin * KH * KW;
    constexpr int khw   = KH * KW;

    const int b    = blockIdx.x;
    const int tid  = threadIdx.x;
    const int lane = tid & 31;
    const int wave = tid >> 5;
    const int hsel = lane >> 4;

    // ---- stage image into LDS (one HBM read per element) ------------------
    const _Float16* gin = in + (size_t)b * IMG;
#if HAVE_ASYNC_LDS
    {
        typedef __attribute__((address_space(1))) vi4 gvi4;
        typedef __attribute__((address_space(3))) vi4 lvi4;
        for (int i = tid; i < IMG / 8; i += 256) {
            __builtin_amdgcn_global_load_async_to_lds_b128(
                (gvi4*)(gin + i * 8), (lvi4*)(sh + i * 8), 0, 0);
        }
        asm volatile("s_wait_asynccnt 0x0" ::: "memory");
    }
#else
    for (int i = tid; i < IMG / 8; i += 256)
        ((uint4*)sh)[i] = ((const uint4*)gin)[i];
#endif
    __syncthreads();

    const _Float16* pwl = pw + (size_t)lane * 16;

    // ---- each wave sweeps n-tiles of this image ----------------------------
    for (int tile = wave; tile < TILES; tile += 8) {     // wave-uniform bounds
        const int nl    = tile * 16 + (lane & 15);       // local position id
        const bool lval = (nl < P);
        const int ox = nl % Wout;                        // compile-time divisors
        const int oy = nl / Wout;

        v8f acc[MT];
        #pragma unroll
        for (int mt = 0; mt < MT; ++mt) acc[mt] = (v8f){};

        for (int s = 0; s < KSTEPS; ++s) {
            // gather B operand from LDS (lane n = lane%16, k = 8h+e (e<8), 16+8h+(e-8) (e>=8))
            v16h bm;
            const int kbase = s * 32 + hsel * 8;
            #pragma unroll
            for (int e = 0; e < 16; ++e) {
                const int kk = kbase + e + (e >= 8 ? 8 : 0);
                _Float16 v = (_Float16)0.f;
                if (kk < K) {                 // statically pruned when K%32==0
                    const int ci  = kk / khw;
                    const int rem = kk - ci * khw;
                    const int ky  = rem / KW;
                    const int kx  = rem - ky * KW;
                    const int iy  = oy * STR - PAD + ky;
                    const int ix  = ox * STR - PAD + kx;
                    if (iy >= 0 && iy < Hin && ix >= 0 && ix < Win)
                        v = sh[(ci * Hin + iy) * Win + ix];   // ds_load_u16
                }
                bm[e] = v;
            }
            // MT back-to-back WMMAs reuse the gathered B
            #pragma unroll
            for (int mt = 0; mt < MT; ++mt) {
                const _Float16* pa = pwl + (size_t)(mt * KSTEPS + s) * 512;
                __builtin_prefetch(pa + 512, 0, 3);   // next k-step's A block
                v16h a = *(const v16h*)pa;            // 2x global_load_b128
                acc[mt] = __builtin_amdgcn_wmma_f32_16x16x32_f16(
                              false, a, false, bm, (short)0, acc[mt], false, false);
            }
        }

        if (lval) {
            #pragma unroll
            for (int mt = 0; mt < MT; ++mt) {
                #pragma unroll
                for (int r = 0; r < 8; ++r) {
                    const int co = mt * 16 + hsel * 8 + r;   // C/D: m = 8h + r
                    float v = acc[mt][r] + bias[co];
                    v = v > 0.f ? v : 0.f;                   // fused ReLU
                    out[(((long)b * Cout + co) * Hout + oy) * Wout + ox] = (_Float16)v;
                }
            }
        }
    }
}

// ---------------------------------------------------------------------------
// 4) MaxPool (VALID), input already ReLU'd f16
// ---------------------------------------------------------------------------
template <int C, int Hi, int Wi, int Ho, int Wo, int KP, int SP>
__global__ void maxpool_kernel(const _Float16* __restrict__ in, _Float16* __restrict__ out,
                               int N) {
    int idx = blockIdx.x * blockDim.x + threadIdx.x;
    if (idx >= N) return;
    const int ox = idx % Wo;
    int t1 = idx / Wo;
    const int oy = t1 % Ho;
    t1 /= Ho;
    const int c = t1 % C;
    const int b = t1 / C;
    const long base = ((long)b * C + c) * Hi * Wi;
    float m = -3.4e38f;
    #pragma unroll
    for (int ky = 0; ky < KP; ++ky)
        #pragma unroll
        for (int kx = 0; kx < KP; ++kx) {
            float v = (float)in[base + (long)(oy * SP + ky) * Wi + (ox * SP + kx)];
            m = v > m ? v : m;
        }
    out[idx] = (_Float16)m;
}

// ---------------------------------------------------------------------------
// 5) Global average over 7x7 -> f32
// ---------------------------------------------------------------------------
__global__ void avg_kernel(const _Float16* __restrict__ in, float* __restrict__ out, int N) {
    int idx = blockIdx.x * blockDim.x + threadIdx.x;   // (b,c) pairs
    if (idx >= N) return;
    const _Float16* p = in + (size_t)idx * 49;
    float s = 0.f;
    #pragma unroll
    for (int i = 0; i < 49; ++i) s += (float)p[i];
    out[idx] = s * (1.0f / 49.0f);
}

// ---------------------------------------------------------------------------
// 6) FC layers (tiny: ~17 MFLOP total)
// ---------------------------------------------------------------------------
__global__ void fc1_kernel(const float* __restrict__ h, const float* __restrict__ w,
                           const float* __restrict__ bias, float* __restrict__ out, int N) {
    int idx = blockIdx.x * blockDim.x + threadIdx.x;
    if (idx >= N) return;
    const int j = idx & 127;
    const int b = idx >> 7;
    const float* hp = h + (size_t)b * 128;
    const float* wp = w + (size_t)j * 128;
    float s = bias[j];
    #pragma unroll 8
    for (int i = 0; i < 128; ++i) s += hp[i] * wp[i];
    out[idx] = s > 0.f ? s : 0.f;
}

__global__ void fc2_kernel(const float* __restrict__ h, const float* __restrict__ w,
                           const float* __restrict__ bias, float* __restrict__ out, int N) {
    int idx = blockIdx.x * blockDim.x + threadIdx.x;
    if (idx >= N) return;
    const int o = idx % OUTc;
    const int b = idx / OUTc;
    const float* hp = h + (size_t)b * 128;
    const float* wp = w + (size_t)o * 128;
    float s = bias[o];
    #pragma unroll 8
    for (int i = 0; i < 128; ++i) s += hp[i] * wp[i];
    out[idx] = s;
}

// ---------------------------------------------------------------------------
// Host-side orchestration
// ---------------------------------------------------------------------------
static inline int cdiv(int a, int b) { return (a + b - 1) / b; }

extern "C" void kernel_launch(void* const* d_in, const int* in_sizes, int n_in,
                              void* d_out, int out_size, void* d_ws, size_t ws_size,
                              hipStream_t stream) {
    const int*   x   = (const int*)  d_in[0];
    const float* w1  = (const float*)d_in[1];
    const float* b1  = (const float*)d_in[2];
    const float* w2  = (const float*)d_in[3];
    const float* b2  = (const float*)d_in[4];
    const float* w3  = (const float*)d_in[5];
    const float* b3  = (const float*)d_in[6];
    const float* fw1 = (const float*)d_in[7];
    const float* fb1 = (const float*)d_in[8];
    const float* fw2 = (const float*)d_in[9];
    const float* fb2 = (const float*)d_in[10];
    float* out = (float*)d_out;

    char* ws = (char*)d_ws;
    const size_t MB = (size_t)1 << 20;
    // Buffer timeline with reuse (peak ~229 MB):
    _Float16* t   = (_Float16*)(ws + 0);          //  64.0 MB [step 1..3]
    _Float16* c1  = (_Float16*)(ws + 64  * MB);   // 128.0 MB [step 3..4]
    _Float16* p1  = (_Float16*)(ws + 192 * MB);   //  30.1 MB [step 4..5]
    _Float16* c2  = (_Float16*)(ws + 0);          //  60.1 MB reuse t [5..6]
    _Float16* p2  = (_Float16*)(ws + 64  * MB);   //  14.1 MB reuse c1 [6..7]
    _Float16* c3  = (_Float16*)(ws + 80  * MB);   //  28.2 MB reuse c1 [7..8]
    _Float16* p3  = (_Float16*)(ws + 112 * MB);   //   6.2 MB reuse c1 [8..9]
    _Float16* pw1 = (_Float16*)(ws + 224 * MB);
    _Float16* pw2 = (_Float16*)(ws + 225 * MB);
    _Float16* pw3 = (_Float16*)(ws + 226 * MB);
    float*    hav = (float*)   (ws + 227 * MB);   // [512,128] f32
    float*    h1  = (float*)   (ws + 228 * MB);   // [512,128] f32
    (void)ws_size; (void)in_sizes; (void)n_in; (void)out_size;

    // 1) rasterize -> t (f16, values exact in f16)
    raster_kernel<<<Bn * Fch, 256, 0, stream>>>(x, t);

    // 2) pack weights into WMMA A layouts
    // L1: Cout=32 K=196 Ksteps=7 mTiles=2 ; L2: 64/800/25/4 ; L3: 128/576/18/8
    pack_weights<<<cdiv(2 * 7 * 512, 256),  256, 0, stream>>>(w1, pw1, 32, 196, 7, 2);
    pack_weights<<<cdiv(4 * 25 * 512, 256), 256, 0, stream>>>(w2, pw2, 64, 800, 25, 4);
    pack_weights<<<cdiv(8 * 18 * 512, 256), 256, 0, stream>>>(w3, pw3, 128, 576, 18, 8);

    // 3) conv1: 4x128x128 -> 32x64x64 (7x7 s2 p3); LDS image = 128 KB
    conv_wmma<4, 128, 128, 32, 64, 64, 7, 7, 2, 3, 7, 2>
        <<<Bn, 256, 4 * 128 * 128 * 2, stream>>>(t, pw1, b1, c1);
    // 4) pool1: 3x3 s2 : 64 -> 31
    {
        int N = Bn * 32 * 31 * 31;
        maxpool_kernel<32, 64, 64, 31, 31, 3, 2>
            <<<cdiv(N, 256), 256, 0, stream>>>(c1, p1, N);
    }
    // 5) conv2: 32x31x31 -> 64x31x31 (5x5 s1 p2); LDS image = 60 KB
    conv_wmma<32, 31, 31, 64, 31, 31, 5, 5, 1, 2, 25, 4>
        <<<Bn, 256, 32 * 31 * 31 * 2, stream>>>(p1, pw2, b2, c2);
    // 6) pool2: 2x2 s2 : 31 -> 15
    {
        int N = Bn * 64 * 15 * 15;
        maxpool_kernel<64, 31, 31, 15, 15, 2, 2>
            <<<cdiv(N, 256), 256, 0, stream>>>(c2, p2, N);
    }
    // 7) conv3: 64x15x15 -> 128x15x15 (3x3 s1 p1); LDS image = 28.1 KB
    conv_wmma<64, 15, 15, 128, 15, 15, 3, 3, 1, 1, 18, 8>
        <<<Bn, 256, 64 * 15 * 15 * 2, stream>>>(p2, pw3, b3, c3);
    // 8) pool3: 2x2 s2 : 15 -> 7
    {
        int N = Bn * 128 * 7 * 7;
        maxpool_kernel<128, 15, 15, 7, 7, 2, 2>
            <<<cdiv(N, 256), 256, 0, stream>>>(c3, p3, N);
    }
    // 9) global average -> [512,128] f32
    avg_kernel<<<cdiv(Bn * 128, 256), 256, 0, stream>>>(p3, hav, Bn * 128);
    // 10) FC1 + ReLU
    fc1_kernel<<<cdiv(Bn * 128, 256), 256, 0, stream>>>(hav, fw1, fb1, h1, Bn * 128);
    // 11) FC2 -> d_out [512,5]
    fc2_kernel<<<cdiv(Bn * OUTc, 256), 256, 0, stream>>>(h1, fw2, fb2, out, Bn * OUTc);
}
